// SAGE_16587163697542
// MI455X (gfx1250) — compile-verified, matrix operand order checked by reading, not run on hardware
//
#include <hip/hip_runtime.h>
#include <hip/hip_bf16.h>
#include <stdint.h>

// ---------------------------------------------------------------------------
// GraphSAGE forward (4 layers) for MI455X / gfx1250.
// Aggregations: f32 atomics (L2-resident, bandwidth/atomic bound).
// GEMMs: bf16 WMMA (v_wmma_f32_16x16x32_bf16), f32 accumulate.
// A/B operands pre-converted to bf16 so the GEMM inner loop is
// pure global_load_b128 + v_wmma; epilogue is branch-free on full tiles.
// ---------------------------------------------------------------------------

typedef __attribute__((ext_vector_type(16))) __bf16 v16bf;
typedef __attribute__((ext_vector_type(8)))  float  v8f;

union FragBF { uint32_t u[8]; v16bf v; };

__device__ __forceinline__ uint32_t cvt2_bf16(float lo, float hi) {
  union { __bf16 b[2]; uint32_t u; } p;
  p.b[0] = (__bf16)lo;
  p.b[1] = (__bf16)hi;
  return p.u;
}

// ---------------------------------------------------------------------------

__global__ void zero_f32(float* __restrict__ p, int n) {
  int i = blockIdx.x * blockDim.x + threadIdx.x;
  if (i < n) p[i] = 0.0f;
}

__global__ void degree_kernel(const int* __restrict__ dst, float* __restrict__ deg, int E) {
  int e = blockIdx.x * blockDim.x + threadIdx.x;
  if (e < E) atomicAdd(&deg[dst[e]], 1.0f);
}

// One wave32 per row: lane holds 4 contiguous floats, reduce sum-of-squares.
__global__ void l2norm_kernel(const float* __restrict__ x, float* __restrict__ h, int n) {
  int gw   = (blockIdx.x * blockDim.x + threadIdx.x) >> 5;
  int lane = threadIdx.x & 31;
  if (gw >= n) return;
  const float4 v = ((const float4*)(x + (long long)gw * 128))[lane];
  float s = v.x * v.x + v.y * v.y + v.z * v.z + v.w * v.w;
#pragma unroll
  for (int m = 16; m >= 1; m >>= 1) s += __shfl_xor(s, m, 32);
  float scale = 1.0f / fmaxf(sqrtf(s), 1e-12f);
  float4 o = { v.x * scale, v.y * scale, v.z * scale, v.w * scale };
  ((float4*)(h + (long long)gw * 128))[lane] = o;
}

// One wave32 per edge: gather h[src] row (float4/lane), atomic-add into agg[dst].
__global__ void scatter_add_kernel(const float* __restrict__ h, const int* __restrict__ src,
                                   const int* __restrict__ dst, float* __restrict__ agg, int E) {
  int gw   = (blockIdx.x * blockDim.x + threadIdx.x) >> 5;
  int lane = threadIdx.x & 31;
  if (gw >= E) return;
  int s = src[gw], d = dst[gw];
  float4 v = ((const float4*)(h + (long long)s * 128))[lane];
  float* base = agg + (long long)d * 128 + lane * 4;
  atomicAdd(base + 0, v.x);
  atomicAdd(base + 1, v.y);
  atomicAdd(base + 2, v.z);
  atomicAdd(base + 3, v.w);
}

// Mean-layer operand prep: hb = bf16(h), nb = bf16(agg / max(deg,1)), row-major bf16.
// One thread per 4 elements (32 threads per 128-wide row).
__global__ void prep_mean_kernel(const float* __restrict__ h, const float* __restrict__ agg,
                                 const float* __restrict__ deg,
                                 uint32_t* __restrict__ hb, uint32_t* __restrict__ nb, int n) {
  int t = blockIdx.x * blockDim.x + threadIdx.x;
  if (t >= n * 32) return;
  int row = t >> 5;
  float invd = 1.0f / fmaxf(deg[row], 1.0f);
  float4 v = ((const float4*)h)[t];
  float4 a = ((const float4*)agg)[t];
  uint2 hp = { cvt2_bf16(v.x, v.y), cvt2_bf16(v.z, v.w) };
  uint2 np = { cvt2_bf16(a.x * invd, a.y * invd), cvt2_bf16(a.z * invd, a.w * invd) };
  ((uint2*)hb)[t] = hp;
  ((uint2*)nb)[t] = np;
}

// GCN-layer operand prep: gb = bf16((h + agg) / (deg + 1)), row-major bf16.
__global__ void prep_gcn_kernel(const float* __restrict__ h, const float* __restrict__ agg,
                                const float* __restrict__ deg,
                                uint32_t* __restrict__ gb, int n) {
  int t = blockIdx.x * blockDim.x + threadIdx.x;
  if (t >= n * 32) return;
  int row = t >> 5;
  float invd = 1.0f / (deg[row] + 1.0f);
  float4 v = ((const float4*)h)[t];
  float4 a = ((const float4*)agg)[t];
  uint2 gp = { cvt2_bf16((v.x + a.x) * invd, (v.y + a.y) * invd),
               cvt2_bf16((v.z + a.z) * invd, (v.w + a.w) * invd) };
  ((uint2*)gb)[t] = gp;
}

// Pre-pack W [K x Nf] (f32 row-major) into CDNA5 WMMA B-fragment layout:
// out[((ntile*(K/32)+ks)*32 + lane)*8 + j] = pack(bf16(W[k0][n]), bf16(W[k0+1][n]))
// k0 = ks*32 + ((j&4)?16:0) + (j&3)*2 + (lane>>4)*8 ; n = ntile*16 + (lane&15)
__global__ void pack_w_kernel(const float* __restrict__ W, uint32_t* __restrict__ out,
                              int K, int Nf) {
  int idx = blockIdx.x * blockDim.x + threadIdx.x;
  int ksteps = K >> 5;
  int total  = (Nf >> 4) * ksteps * 256;
  if (idx >= total) return;
  int j     = idx & 7;
  int lane  = (idx >> 3) & 31;
  int rest  = idx >> 8;
  int ks    = rest % ksteps;
  int ntile = rest / ksteps;
  int half  = lane >> 4;
  int k0    = (ks << 5) + ((j & 4) << 2) + ((j & 3) << 1) + (half << 3);
  int ncol  = (ntile << 4) + (lane & 15);
  out[idx]  = cvt2_bf16(W[k0 * Nf + ncol], W[(k0 + 1) * Nf + ncol]);
}

// ---------------------------------------------------------------------------
// A-fragment loader from a row-major bf16 row (viewed as uint32 pairs,
// 64 dwords per row). For k-step ks, lane-half h: dwords [ks*16+h*4 .. +3]
// and [ks*16+h*4+8 .. +11] form the 16x32 bf16 A fragment per ISA layout.
// ---------------------------------------------------------------------------
__device__ __forceinline__ void load_fragA(FragBF& a, const uint32_t* row, int abase) {
  *(uint4*)&a.u[0] = *(const uint4*)(row + abase);
  *(uint4*)&a.u[4] = *(const uint4*)(row + abase + 8);
}
__device__ __forceinline__ void load_fragB(FragBF& b, const uint32_t* frag) {
  *(uint4*)&b.u[0] = *(const uint4*)(frag);
  *(uint4*)&b.u[4] = *(const uint4*)(frag + 4);
}

// GEMM (mean layers): out = relu?(Hb @ Wself + Nb @ Wneigh + bias)
// One wave32 per 16x16 output tile; K = 128 -> 4 WMMA steps per weight matrix.
template <int NFOUT, bool RELU>
__global__ void gemm_dual_kernel(const uint32_t* __restrict__ Hb, const uint32_t* __restrict__ Nb,
                                 const uint32_t* __restrict__ Wsp, const uint32_t* __restrict__ Wnp,
                                 const float* __restrict__ bias, float* __restrict__ out, int n) {
  constexpr int TN = NFOUT / 16;
  int gw   = (blockIdx.x * blockDim.x + threadIdx.x) >> 5;
  int lane = threadIdx.x & 31;
  int mtile = gw / TN;
  int ntile = gw % TN;
  if (mtile * 16 >= n) return;                 // wave-uniform: EXEC stays all-ones
  int half = lane >> 4;
  int m    = lane & 15;
  int r    = mtile * 16 + m; if (r > n - 1) r = n - 1;
  const uint32_t* hrow = Hb + (long long)r * 64;
  const uint32_t* nrow = Nb + (long long)r * 64;
  float bv = bias[ntile * 16 + m];
  v8f c = { bv, bv, bv, bv, bv, bv, bv, bv };
#pragma unroll
  for (int ks = 0; ks < 4; ++ks) {
    int abase = ks * 16 + half * 4;
    FragBF a, b;
    load_fragA(a, hrow, abase);
    load_fragB(b, Wsp + ((ntile * 4 + ks) * 256 + lane * 8));
    c = __builtin_amdgcn_wmma_f32_16x16x32_bf16(false, a.v, false, b.v, (short)0, c, false, false);
    load_fragA(a, nrow, abase);
    load_fragB(b, Wnp + ((ntile * 4 + ks) * 256 + lane * 8));
    c = __builtin_amdgcn_wmma_f32_16x16x32_bf16(false, a.v, false, b.v, (short)0, c, false, false);
  }
  // Epilogue: single wave-uniform bounds test; branch-free stores on full tiles.
  float* op = out + (long long)(mtile * 16 + half * 8) * NFOUT + ntile * 16 + m;
  if (mtile * 16 + 16 <= n) {
#pragma unroll
    for (int i = 0; i < 8; ++i) {
      float v = c[i];
      if (RELU) v = fmaxf(v, 0.0f);
      op[(long long)i * NFOUT] = v;
    }
  } else {
#pragma unroll
    for (int i = 0; i < 8; ++i) {
      int ro = mtile * 16 + i + half * 8;
      float v = c[i];
      if (RELU) v = fmaxf(v, 0.0f);
      if (ro < n) op[(long long)i * NFOUT] = v;
    }
  }
}

// GEMM (GCN layers): out = relu(Gb @ Wneigh + bias), 128-wide output.
__global__ void gemm_single_kernel(const uint32_t* __restrict__ Gb,
                                   const uint32_t* __restrict__ Wp,
                                   const float* __restrict__ bias, float* __restrict__ out, int n) {
  int gw   = (blockIdx.x * blockDim.x + threadIdx.x) >> 5;
  int lane = threadIdx.x & 31;
  int mtile = gw >> 3;       // 128/16 = 8 n-tiles
  int ntile = gw & 7;
  if (mtile * 16 >= n) return;
  int half = lane >> 4;
  int m    = lane & 15;
  int r    = mtile * 16 + m; if (r > n - 1) r = n - 1;
  const uint32_t* grow = Gb + (long long)r * 64;
  float bv = bias[ntile * 16 + m];
  v8f c = { bv, bv, bv, bv, bv, bv, bv, bv };
#pragma unroll
  for (int ks = 0; ks < 4; ++ks) {
    FragBF a, b;
    load_fragA(a, grow, ks * 16 + half * 4);
    load_fragB(b, Wp + ((ntile * 4 + ks) * 256 + lane * 8));
    c = __builtin_amdgcn_wmma_f32_16x16x32_bf16(false, a.v, false, b.v, (short)0, c, false, false);
  }
  float* op = out + (long long)(mtile * 16 + half * 8) * 128 + ntile * 16 + m;
  if (mtile * 16 + 16 <= n) {
#pragma unroll
    for (int i = 0; i < 8; ++i) op[(long long)i * 128] = fmaxf(c[i], 0.0f);
  } else {
#pragma unroll
    for (int i = 0; i < 8; ++i) {
      int ro = mtile * 16 + i + half * 8;
      if (ro < n) op[(long long)i * 128] = fmaxf(c[i], 0.0f);
    }
  }
}

// ---------------------------------------------------------------------------

extern "C" void kernel_launch(void* const* d_in, const int* in_sizes, int n_in,
                              void* d_out, int out_size, void* d_ws, size_t ws_size,
                              hipStream_t stream) {
  const float* x   = (const float*)d_in[0];
  const int*   src = (const int*)d_in[1];
  const int*   dst = (const int*)d_in[2];
  const float* Ws0 = (const float*)d_in[3];
  const float* Wn0 = (const float*)d_in[4];
  const float* b0  = (const float*)d_in[5];
  const float* Wn1 = (const float*)d_in[6];
  const float* b1  = (const float*)d_in[7];
  const float* Wn2 = (const float*)d_in[8];
  const float* b2  = (const float*)d_in[9];
  const float* Ws3 = (const float*)d_in[10];
  const float* Wn3 = (const float*)d_in[11];
  const float* b3  = (const float*)d_in[12];

  const int n = in_sizes[0] / 128;
  const int E = in_sizes[1];

  // Workspace carve-out
  char*  ws  = (char*)d_ws;
  size_t off = 0;
  auto carve = [&](size_t bytes) -> void* {
    void* p = ws + off;
    off = (off + bytes + 255) & ~(size_t)255;
    return p;
  };
  float*    degv = (float*)carve((size_t)n * sizeof(float));
  float*    hA   = (float*)carve((size_t)n * 128 * sizeof(float));
  float*    hB   = (float*)carve((size_t)n * 128 * sizeof(float));
  float*    agg  = (float*)carve((size_t)n * 128 * sizeof(float));
  uint32_t* bA   = (uint32_t*)carve((size_t)n * 64 * sizeof(uint32_t)); // bf16 NxF
  uint32_t* bN   = (uint32_t*)carve((size_t)n * 64 * sizeof(uint32_t)); // bf16 NxF
  uint32_t* pS   = (uint32_t*)carve(8192 * sizeof(uint32_t));
  uint32_t* pN   = (uint32_t*)carve(8192 * sizeof(uint32_t));

  const int B = 256;
  const int nfeat     = n * 128;
  const int zeroBlksF = (nfeat + B - 1) / B;
  const int zeroBlksN = (n + B - 1) / B;
  const int edgeBlks  = (E + B - 1) / B;
  const int waveBlksE = ((E * 32) + B - 1) / B;          // one wave per edge
  const int waveBlksN = ((n * 32) + B - 1) / B;          // one wave per row
  const int prepBlks  = ((n * 32) + B - 1) / B;          // thread per 4 elems
  const int tilesM    = (n + 15) / 16;
  const int gemmBlksH = ((tilesM * 8 * 32) + B - 1) / B; // 8 n-tiles (128 out)
  const int gemmBlksO = ((tilesM * 2 * 32) + B - 1) / B; // 2 n-tiles (32 out)

  // Degree (computed once)
  zero_f32<<<zeroBlksN, B, 0, stream>>>(degv, n);
  degree_kernel<<<edgeBlks, B, 0, stream>>>(dst, degv, E);

  // h = l2_normalize(x)
  l2norm_kernel<<<waveBlksN, B, 0, stream>>>(x, hA, n);

  // ---- Layer 0 (mean): hB = relu(hA@Ws0 + (sum_nbr hA / max(deg,1))@Wn0 + b0)
  zero_f32<<<zeroBlksF, B, 0, stream>>>(agg, nfeat);
  scatter_add_kernel<<<waveBlksE, B, 0, stream>>>(hA, src, dst, agg, E);
  prep_mean_kernel<<<prepBlks, B, 0, stream>>>(hA, agg, degv, bA, bN, n);
  pack_w_kernel<<<(8192 + B - 1) / B, B, 0, stream>>>(Ws0, pS, 128, 128);
  pack_w_kernel<<<(8192 + B - 1) / B, B, 0, stream>>>(Wn0, pN, 128, 128);
  gemm_dual_kernel<128, true><<<gemmBlksH, B, 0, stream>>>(bA, bN, pS, pN, b0, hB, n);

  // ---- Layer 1 (gcn): hA = relu(((sum_nbr hB + hB)/(deg+1))@Wn1 + b1)
  zero_f32<<<zeroBlksF, B, 0, stream>>>(agg, nfeat);
  scatter_add_kernel<<<waveBlksE, B, 0, stream>>>(hB, src, dst, agg, E);
  prep_gcn_kernel<<<prepBlks, B, 0, stream>>>(hB, agg, degv, bA, n);
  pack_w_kernel<<<(8192 + B - 1) / B, B, 0, stream>>>(Wn1, pS, 128, 128);
  gemm_single_kernel<<<gemmBlksH, B, 0, stream>>>(bA, pS, b1, hA, n);

  // ---- Layer 2 (gcn): hB = relu(((sum_nbr hA + hA)/(deg+1))@Wn2 + b2)
  zero_f32<<<zeroBlksF, B, 0, stream>>>(agg, nfeat);
  scatter_add_kernel<<<waveBlksE, B, 0, stream>>>(hA, src, dst, agg, E);
  prep_gcn_kernel<<<prepBlks, B, 0, stream>>>(hA, agg, degv, bA, n);
  pack_w_kernel<<<(8192 + B - 1) / B, B, 0, stream>>>(Wn2, pS, 128, 128);
  gemm_single_kernel<<<gemmBlksH, B, 0, stream>>>(bA, pS, b2, hB, n);

  // ---- Layer 3 (mean, no relu): out = hB@Ws3 + (sum_nbr hB / max(deg,1))@Wn3 + b3
  zero_f32<<<zeroBlksF, B, 0, stream>>>(agg, nfeat);
  scatter_add_kernel<<<waveBlksE, B, 0, stream>>>(hB, src, dst, agg, E);
  prep_mean_kernel<<<prepBlks, B, 0, stream>>>(hB, agg, degv, bA, bN, n);
  pack_w_kernel<<<(2048 + B - 1) / B, B, 0, stream>>>(Ws3, pS, 128, 32);
  pack_w_kernel<<<(2048 + B - 1) / B, B, 0, stream>>>(Wn3, pN, 128, 32);
  gemm_dual_kernel<32, false><<<gemmBlksO, B, 0, stream>>>(bA, bN, pS, pN, b3,
                                                           (float*)d_out, n);
}
